// MutiHeadAttention_17944373362882
// MI455X (gfx1250) — matrix operand order repaired and checked by reading, not compiled, not run on hardware
//
#include <hip/hip_runtime.h>

// MHA forward, B=4 S=2048 D=1024 H=16 HD=64, causal, fp32 in/out.
// Pipeline: (1) split x -> f16 hi/lo, weights -> transposed f16 hi/lo
//           (2) QKV GEMM (f16x3 WMMA, 32x64 wave tiles), q/k [b,h,s,hd], v^T [b,h,hd,s]
//           (3) flash attention (f16 WMMA, online softmax), ctx -> f16 hi/lo
//           (4) output GEMM (f16x3 WMMA, 32x64 wave tiles) + bias -> fp32 d_out
// Workspace: ~128 MiB (all L2-resident on MI455X's 192MB L2).

#define BB   4
#define SS   2048
#define DIN  1024
#define DOUT 1024
#define HH   16
#define HD   64

typedef __attribute__((ext_vector_type(16))) _Float16 v16h;
typedef __attribute__((ext_vector_type(8)))  _Float16 v8h;
typedef __attribute__((ext_vector_type(8)))  float    v8f;

// ---- WMMA helpers -----------------------------------------------------------

__device__ __forceinline__ v8f wmma_f16(v16h a, v16h b, v8f c) {
  // D = A(16x32 f16) * B(32x16 f16) + C(16x16 f32)
  return __builtin_amdgcn_wmma_f32_16x16x32_f16(
      /*neg_a=*/false, a, /*neg_b=*/false, b,
      /*c_mod=*/(short)0, c, /*reuse_a=*/false, /*reuse_b=*/false);
}

// A fragment (16x32, 16-bit): lane<16 holds row=lane, K = k0+[0..7] and k0+16+[0..7];
// lane>=16 holds row=lane-16, K = k0+8+[0..7] and k0+24+[0..7].
__device__ __forceinline__ v16h load_a_frag(const _Float16* base, int ld,
                                            int mrow, int half, int k0) {
  union { v16h v; v8h h[2]; } u;
  const _Float16* p = base + (size_t)mrow * ld + k0 + half * 8;
  u.h[0] = *(const v8h*)(p);
  u.h[1] = *(const v8h*)(p + 16);
  return u.v;
}

// B fragment (32x16, 16-bit), source row-major [N][K] (i.e. transposed weight):
// lane<16 holds col N=lane, K=k0+[0..7] and k0+16+[0..7]; lane>=16 holds N=lane-16, K=k0+8+[0..7] and k0+24+[0..7].
__device__ __forceinline__ v16h load_b_frag(const _Float16* btNK, int ld,
                                            int ncol, int half, int k0) {
  union { v16h v; v8h h[2]; } u;
  const _Float16* p = btNK + (size_t)ncol * ld + k0 + half * 8;
  u.h[0] = *(const v8h*)(p);
  u.h[1] = *(const v8h*)(p + 16);
  return u.v;
}

// ---- precision split kernels ------------------------------------------------

__global__ void split_rowmajor(const float* __restrict__ in,
                               _Float16* __restrict__ hi, _Float16* __restrict__ lo,
                               int n) {
  int i = blockIdx.x * blockDim.x + threadIdx.x;
  if (i < n) {
    float v = in[i];
    _Float16 h = (_Float16)v;
    hi[i] = h;
    lo[i] = (_Float16)(v - (float)h);
  }
}

// in [K][N] fp32 -> hi/lo f16 [N][K] (transposed for contiguous B-fragments)
__global__ void split_transpose(const float* __restrict__ in,
                                _Float16* __restrict__ hi, _Float16* __restrict__ lo,
                                int K, int N) {
  int i = blockIdx.x * blockDim.x + threadIdx.x;
  if (i < K * N) {
    int kk = i / N, nn = i - kk * N;
    float v = in[i];
    _Float16 h = (_Float16)v;
    size_t o = (size_t)nn * K + kk;
    hi[o] = h;
    lo[o] = (_Float16)(v - (float)h);
  }
}

// ---- QKV projection: y = x @ W, f16x3, head-split epilogue ------------------
// block = 8 waves, wave tile 32(M) x 64(N); grid (M/256, N/64, 3)

__global__ __launch_bounds__(256) void qkv_gemm(
    const _Float16* __restrict__ xh, const _Float16* __restrict__ xl,
    const _Float16* __restrict__ wqh, const _Float16* __restrict__ wql,
    const _Float16* __restrict__ wkh, const _Float16* __restrict__ wkl,
    const _Float16* __restrict__ wvh, const _Float16* __restrict__ wvl,
    _Float16* __restrict__ qout, _Float16* __restrict__ kout,
    _Float16* __restrict__ vtout) {
  const int wave = threadIdx.x >> 5;
  const int lane = threadIdx.x & 31;
  const int half = lane >> 4;
  const int lidx = lane & 15;
  const int m0 = blockIdx.x * 256 + wave * 32;
  const int n0 = blockIdx.y * 64;
  const int which = blockIdx.z;

  const _Float16* wth = (which == 0) ? wqh : (which == 1) ? wkh : wvh;
  const _Float16* wtl = (which == 0) ? wql : (which == 1) ? wkl : wvl;

  v8f acc[2][4];
#pragma unroll
  for (int mi = 0; mi < 2; ++mi)
#pragma unroll
    for (int j = 0; j < 4; ++j) acc[mi][j] = (v8f)(0.0f);

  const _Float16* ah0p = xh + (size_t)m0 * DIN;
  const _Float16* ah1p = xh + (size_t)(m0 + 16) * DIN;
  const _Float16* al0p = xl + (size_t)m0 * DIN;
  const _Float16* al1p = xl + (size_t)(m0 + 16) * DIN;

  for (int k0 = 0; k0 < DIN; k0 += 32) {
    v16h ah0 = load_a_frag(ah0p, DIN, lidx, half, k0);
    v16h ah1 = load_a_frag(ah1p, DIN, lidx, half, k0);
    v16h al0 = load_a_frag(al0p, DIN, lidx, half, k0);
    v16h al1 = load_a_frag(al1p, DIN, lidx, half, k0);
#pragma unroll
    for (int j = 0; j < 4; ++j) {
      v16h bh = load_b_frag(wth + (size_t)(n0 + j * 16) * DIN, DIN, lidx, half, k0);
      v16h bl = load_b_frag(wtl + (size_t)(n0 + j * 16) * DIN, DIN, lidx, half, k0);
      acc[0][j] = wmma_f16(ah0, bh, acc[0][j]);
      acc[0][j] = wmma_f16(al0, bh, acc[0][j]);
      acc[0][j] = wmma_f16(ah0, bl, acc[0][j]);
      acc[1][j] = wmma_f16(ah1, bh, acc[1][j]);
      acc[1][j] = wmma_f16(al1, bh, acc[1][j]);
      acc[1][j] = wmma_f16(ah1, bl, acc[1][j]);
    }
  }

  // C layout: lane holds N = j*16 + lidx, rows M = 8*half + r
#pragma unroll
  for (int mi = 0; mi < 2; ++mi) {
#pragma unroll
    for (int j = 0; j < 4; ++j) {
      int n = n0 + j * 16 + lidx;
      int h = n >> 6, hd = n & 63;
#pragma unroll
      for (int r = 0; r < 8; ++r) {
        int m = m0 + mi * 16 + 8 * half + r;
        int b = m >> 11, s = m & (SS - 1);
        _Float16 val = (_Float16)acc[mi][j][r];
        if (which == 2) {
          vtout[(((size_t)(b * HH + h)) * HD + hd) * SS + s] = val;  // [b,h,hd,t]
        } else {
          _Float16* o = (which == 0) ? qout : kout;
          o[(((size_t)(b * HH + h)) * SS + s) * HD + hd] = val;       // [b,h,s,hd]
        }
      }
    }
  }
}

// ---- flash attention (causal), block = 4 waves x 16 q-rows ------------------
// grid (S/64, B*H), block 128

__global__ __launch_bounds__(128) void attn_fwd(
    const _Float16* __restrict__ q,   // [B,H,S,HD]
    const _Float16* __restrict__ k,   // [B,H,S,HD]
    const _Float16* __restrict__ vt,  // [B,H,HD,S]
    _Float16* __restrict__ ctxh, _Float16* __restrict__ ctxl) {
  __shared__ __align__(16) _Float16 plds[4][16 * 32];

  const int wave = threadIdx.x >> 5;
  const int lane = threadIdx.x & 31;
  const int half = lane >> 4;
  const int lidx = lane & 15;
  const int bh = blockIdx.y;
  const int qbase = blockIdx.x * 64 + wave * 16;

  const _Float16* qp = q + ((size_t)bh * SS + qbase) * HD;
  const _Float16* kp = k + (size_t)bh * SS * HD;
  const _Float16* vp = vt + (size_t)bh * HD * SS;

  // Q fragments for hd = 0..63 (two K=32 steps), resident for whole kernel
  v16h qa0 = load_a_frag(qp, HD, lidx, half, 0);
  v16h qa1 = load_a_frag(qp, HD, lidx, half, 32);

  v8f o[4];
#pragma unroll
  for (int j = 0; j < 4; ++j) o[j] = (v8f)(0.0f);
  float mrow[8], lrow[8];
#pragma unroll
  for (int r = 0; r < 8; ++r) { mrow[r] = -1e30f; lrow[r] = 0.0f; }

  const float cscale = 1.4426950408889634f * 0.125f;  // log2(e)/sqrt(64)
  const int tend = blockIdx.x * 64 + 63;              // causal bound for block

  for (int t0 = 0; t0 <= tend; t0 += 32) {
    // scores S = Q K^T : two 16x16 C frags covering t0..t0+31
    v8f sc[2];
#pragma unroll
    for (int jj = 0; jj < 2; ++jj) {
      v8f c = (v8f)(0.0f);
      v16h b0 = load_b_frag(kp + (size_t)(t0 + jj * 16) * HD, HD, lidx, half, 0);
      v16h b1 = load_b_frag(kp + (size_t)(t0 + jj * 16) * HD, HD, lidx, half, 32);
      c = wmma_f16(qa0, b0, c);
      c = wmma_f16(qa1, b1, c);
      sc[jj] = c;
    }

    // online softmax in C-layout (lane = column t, vgpr r + 8*half = row)
#pragma unroll
    for (int r = 0; r < 8; ++r) {
      int qg = qbase + 8 * half + r;
      float mx = -1e30f;
#pragma unroll
      for (int jj = 0; jj < 2; ++jj) {
        int tg = t0 + jj * 16 + lidx;
        float s = sc[jj][r] * cscale;
        s = (tg <= qg) ? s : -1e30f;   // causal mask
        sc[jj][r] = s;
        mx = fmaxf(mx, s);
      }
#pragma unroll
      for (int off = 1; off < 16; off <<= 1)
        mx = fmaxf(mx, __shfl_xor(mx, off, 32));  // reduce within 16-lane half
      float mnew = fmaxf(mrow[r], mx);
      float alpha = exp2f(mrow[r] - mnew);
      float rsum = 0.0f;
#pragma unroll
      for (int jj = 0; jj < 2; ++jj) {
        float p = exp2f(sc[jj][r] - mnew);
        sc[jj][r] = p;
        rsum += p;
      }
#pragma unroll
      for (int off = 1; off < 16; off <<= 1)
        rsum += __shfl_xor(rsum, off, 32);
      lrow[r] = lrow[r] * alpha + rsum;
      mrow[r] = mnew;
#pragma unroll
      for (int j = 0; j < 4; ++j) o[j][r] *= alpha;
    }

    // C-layout -> A-layout via per-wave LDS tile (16 rows x 32 cols of P)
    _Float16* pw = &plds[wave][0];
#pragma unroll
    for (int jj = 0; jj < 2; ++jj)
#pragma unroll
      for (int r = 0; r < 8; ++r)
        pw[(8 * half + r) * 32 + jj * 16 + lidx] = (_Float16)sc[jj][r];

    union { v16h v; v8h h[2]; } up;
    const _Float16* pr = pw + lidx * 32 + half * 8;
    up.h[0] = *(const v8h*)(pr);
    up.h[1] = *(const v8h*)(pr + 16);

    // O += P @ V   (V pre-transposed: contiguous B-fragments)
#pragma unroll
    for (int j = 0; j < 4; ++j) {
      v16h bv = load_b_frag(vp + (size_t)(j * 16) * SS + t0, SS, lidx, half, 0);
      o[j] = wmma_f16(up.v, bv, o[j]);
    }
  }

  // epilogue: normalize, split fp32 -> f16 hi/lo for the output GEMM
  const int b = bh >> 4, h = bh & 15;
#pragma unroll
  for (int r = 0; r < 8; ++r) {
    float inv = 1.0f / lrow[r];
    int m = b * SS + qbase + 8 * half + r;
#pragma unroll
    for (int j = 0; j < 4; ++j) {
      int n = h * 64 + j * 16 + lidx;
      float val = o[j][r] * inv;
      _Float16 hi = (_Float16)val;
      size_t oidx = (size_t)m * DOUT + n;
      ctxh[oidx] = hi;
      ctxl[oidx] = (_Float16)(val - (float)hi);
    }
  }
}

// ---- output projection: out = ctx @ wo + bo (f16x3, 32x64 wave tiles) -------

__global__ __launch_bounds__(256) void out_gemm(
    const _Float16* __restrict__ ah, const _Float16* __restrict__ al,  // [M,1024]
    const _Float16* __restrict__ bth, const _Float16* __restrict__ btl, // [N][K]
    const float* __restrict__ bias, float* __restrict__ out) {
  const int wave = threadIdx.x >> 5;
  const int lane = threadIdx.x & 31;
  const int half = lane >> 4;
  const int lidx = lane & 15;
  const int m0 = blockIdx.x * 256 + wave * 32;
  const int n0 = blockIdx.y * 64;

  v8f acc[2][4];
#pragma unroll
  for (int mi = 0; mi < 2; ++mi)
#pragma unroll
    for (int j = 0; j < 4; ++j) acc[mi][j] = (v8f)(0.0f);

  const _Float16* ah0p = ah + (size_t)m0 * DOUT;
  const _Float16* ah1p = ah + (size_t)(m0 + 16) * DOUT;
  const _Float16* al0p = al + (size_t)m0 * DOUT;
  const _Float16* al1p = al + (size_t)(m0 + 16) * DOUT;

  for (int k0 = 0; k0 < DOUT; k0 += 32) {
    v16h xh0 = load_a_frag(ah0p, DOUT, lidx, half, k0);
    v16h xh1 = load_a_frag(ah1p, DOUT, lidx, half, k0);
    v16h xl0 = load_a_frag(al0p, DOUT, lidx, half, k0);
    v16h xl1 = load_a_frag(al1p, DOUT, lidx, half, k0);
#pragma unroll
    for (int j = 0; j < 4; ++j) {
      v16h bh = load_b_frag(bth + (size_t)(n0 + j * 16) * DOUT, DOUT, lidx, half, k0);
      v16h bl = load_b_frag(btl + (size_t)(n0 + j * 16) * DOUT, DOUT, lidx, half, k0);
      acc[0][j] = wmma_f16(xh0, bh, acc[0][j]);
      acc[0][j] = wmma_f16(xl0, bh, acc[0][j]);
      acc[0][j] = wmma_f16(xh0, bl, acc[0][j]);
      acc[1][j] = wmma_f16(xh1, bh, acc[1][j]);
      acc[1][j] = wmma_f16(xl1, bh, acc[1][j]);
      acc[1][j] = wmma_f16(xh1, bl, acc[1][j]);
    }
  }

#pragma unroll
  for (int mi = 0; mi < 2; ++mi) {
#pragma unroll
    for (int j = 0; j < 4; ++j) {
      int n = n0 + j * 16 + lidx;
      float bi = bias[n];
#pragma unroll
      for (int r = 0; r < 8; ++r) {
        int m = m0 + mi * 16 + 8 * half + r;
        out[(size_t)m * DOUT + n] = acc[mi][j][r] + bi;
      }
    }
  }
}

// ---- launch -----------------------------------------------------------------

extern "C" void kernel_launch(void* const* d_in, const int* in_sizes, int n_in,
                              void* d_out, int out_size, void* d_ws, size_t ws_size,
                              hipStream_t stream) {
  const float* x  = (const float*)d_in[0];
  const float* wq = (const float*)d_in[1];
  const float* wk = (const float*)d_in[2];
  const float* wv = (const float*)d_in[3];
  const float* wo = (const float*)d_in[4];
  const float* bo = (const float*)d_in[5];
  float* out = (float*)d_out;

  const size_t NX = (size_t)BB * SS * DIN;   // 8,388,608
  const size_t NW = (size_t)DIN * DOUT;      // 1,048,576

  char* ws = (char*)d_ws;
  size_t off = 0;
  auto carve = [&](size_t bytes) -> _Float16* {
    _Float16* p = (_Float16*)(ws + off);
    off += (bytes + 255) & ~(size_t)255;
    return p;
  };
  _Float16* xh   = carve(NX * 2);
  _Float16* xl   = carve(NX * 2);
  _Float16* wqh  = carve(NW * 2);
  _Float16* wql  = carve(NW * 2);
  _Float16* wkh  = carve(NW * 2);
  _Float16* wkl  = carve(NW * 2);
  _Float16* wvh  = carve(NW * 2);
  _Float16* wvl  = carve(NW * 2);
  _Float16* woh  = carve(NW * 2);
  _Float16* wol  = carve(NW * 2);
  _Float16* qb   = carve(NX * 2);
  _Float16* kb   = carve(NX * 2);
  _Float16* vtb  = carve(NX * 2);
  _Float16* ctxh = carve(NX * 2);
  _Float16* ctxl = carve(NX * 2);
  (void)ws_size; // requires ~128 MiB workspace

  const int tb = 256;
  split_rowmajor<<<(unsigned)((NX + tb - 1) / tb), tb, 0, stream>>>(x, xh, xl, (int)NX);
  split_transpose<<<(unsigned)((NW + tb - 1) / tb), tb, 0, stream>>>(wq, wqh, wql, DIN, DOUT);
  split_transpose<<<(unsigned)((NW + tb - 1) / tb), tb, 0, stream>>>(wk, wkh, wkl, DIN, DOUT);
  split_transpose<<<(unsigned)((NW + tb - 1) / tb), tb, 0, stream>>>(wv, wvh, wvl, DIN, DOUT);
  split_transpose<<<(unsigned)((NW + tb - 1) / tb), tb, 0, stream>>>(wo, woh, wol, DOUT, DOUT);

  qkv_gemm<<<dim3((BB * SS) / 256, DOUT / 64, 3), 256, 0, stream>>>(
      xh, xl, wqh, wql, wkh, wkl, wvh, wvl, qb, kb, vtb);

  attn_fwd<<<dim3(SS / 64, BB * HH), 128, 0, stream>>>(qb, kb, vtb, ctxh, ctxl);

  out_gemm<<<dim3((BB * SS) / 256, DOUT / 64), 256, 0, stream>>>(
      ctxh, ctxl, woh, wol, bo, out);
}
